// LowFreqDifferentialAttention_27350351741152
// MI455X (gfx1250) — matrix operand order repaired and checked by reading, not compiled, not run on hardware
//
#include <hip/hip_runtime.h>
#include <hip/hip_bf16.h>
#include <math.h>

typedef _Float16 h16;
typedef __attribute__((ext_vector_type(16))) _Float16 v16h;
typedef __attribute__((ext_vector_type(8)))  _Float16 v8h;
typedef __attribute__((ext_vector_type(8)))  float    v8f;
typedef __attribute__((ext_vector_type(4)))  int      v4i;

#define BB 4
#define CC 64
#define NN 4096      // H*W
#define DD 64
#define HID 256

// ---------------------------------------------------------------------------
// Async global->LDS staging (gfx1250). The builtin takes address-space
// qualified int4 pointers: (as1 int4* gsrc, as3 int4* ldst, imm off, imm cpol).
// Guarded so a toolchain without the builtin still compiles (sync fallback).
// ---------------------------------------------------------------------------
typedef __attribute__((address_space(1))) v4i as1v4i;
typedef __attribute__((address_space(3))) v4i as3v4i;

#if __has_builtin(__builtin_amdgcn_global_load_async_to_lds_b128)
#define HAVE_ASYNC 1
#endif

#if __has_builtin(__builtin_amdgcn_s_wait_asynccnt)
#define WAIT_ASYNC(n) __builtin_amdgcn_s_wait_asynccnt(n)
#else
#define WAIT_ASYNC(n) asm volatile("s_wait_asynccnt %0" ::"n"(n))
#endif

__device__ __forceinline__ void stage_chunk16(h16* ldst, const h16* gsrc) {
#ifdef HAVE_ASYNC
  __builtin_amdgcn_global_load_async_to_lds_b128((as1v4i*)gsrc, (as3v4i*)ldst,
                                                 0, 0);
#else
  *(v8h*)ldst = *(const v8h*)gsrc;   // synchronous fallback
#endif
}

__device__ __forceinline__ void wait_stage(bool next_in_flight) {
#ifdef HAVE_ASYNC
  if (next_in_flight) WAIT_ASYNC(6);   // 6 async ops per wave belong to next tile
  else                WAIT_ASYNC(0);
#else
  (void)next_in_flight;
#endif
}

// ---------------------------------------------------------------------------
// WMMA helpers (layouts per CDNA5 ISA 7.12.2, wave32)
// ---------------------------------------------------------------------------
__device__ __forceinline__ v8f zero8() {
  v8f z = {0.f,0.f,0.f,0.f,0.f,0.f,0.f,0.f};
  return z;
}

__device__ __forceinline__ v8f wmma16(v16h a, v16h b, v8f c) {
  return __builtin_amdgcn_wmma_f32_16x16x32_f16(false, a, false, b, (short)0, c,
                                                false, false);
}

// A fragment: 16x32 f16 tile, row-major ld (halves).
// lanes 0-15: M=lane, K = 0..7 / 16..23 ; lanes 16-31: M=lane-16, K = 8..15 / 24..31
__device__ __forceinline__ v16h load_a_frag(const h16* src, int ld, int lane) {
  int r  = lane & 15;
  int kb = (lane >> 4) * 8;
  const h16* p = src + r * ld + kb;
  v8h lo = *(const v8h*)(p);
  v8h hi = *(const v8h*)(p + 16);
  v16h out;
#pragma unroll
  for (int j = 0; j < 8; ++j) { out[j] = lo[j]; out[j + 8] = hi[j]; }
  return out;
}

// B fragment: 32x16 f16 tile, K contiguous per output column (stride ld).
// lane l holds column l&15, K = (l>=16 ? 8 : 0) + j for j<8, 16 + (l>=16 ? 8 : 0) + (j-8) for j>=8.
__device__ __forceinline__ v16h load_b_frag(const h16* src, int ld, int lane) {
  const h16* p = src + (lane & 15) * ld + (lane >> 4) * 8;
  v8h lo = *(const v8h*)(p);
  v8h hi = *(const v8h*)(p + 16);
  v16h out;
#pragma unroll
  for (int j = 0; j < 8; ++j) { out[j] = lo[j]; out[j + 8] = hi[j]; }
  return out;
}

// ---------------------------------------------------------------------------
// Kernel 0: weight fp32 -> f16 repack. WB = concat(Wq1,Wk1,Wq2,Wk2,Wv) [320][64]
// ---------------------------------------------------------------------------
__global__ void k_convert_weights(const float* Wq1, const float* Wk1,
                                  const float* Wq2, const float* Wk2,
                                  const float* Wv,  const float* Wp,
                                  const float* W1,  const float* W2,
                                  h16* WB, h16* Wph, h16* W1h, h16* W2h) {
  int t = blockIdx.x * blockDim.x + threadIdx.x;
  if (t < 320 * 64) {
    int r = t >> 6, c = t & 63;
    int w = r >> 6, rr = r & 63;
    float v;
    if      (w == 0) v = Wq1[rr * 64 + c];
    else if (w == 1) v = Wk1[rr * 64 + c];
    else if (w == 2) v = Wq2[rr * 64 + c];
    else if (w == 3) v = Wk2[rr * 64 + c];
    else             v = Wv [rr * 64 + c];
    WB[t] = (h16)v;
  }
  if (t < 64 * 64)  Wph[t] = (h16)Wp[t];
  if (t < 256 * 64) { W1h[t] = (h16)W1[t]; W2h[t] = (h16)W2[t]; }
}

// ---------------------------------------------------------------------------
// Kernel 1: Fl = Fs + Ff (fp32, [B][C][N]) and token-major f16 xT [B][N][C]
// ---------------------------------------------------------------------------
__global__ void k_add_transpose(const float* __restrict__ Fs,
                                const float* __restrict__ Ff,
                                float* __restrict__ Fl, h16* __restrict__ xT) {
  __shared__ float tile[64][65];
  int b  = blockIdx.y;
  int n0 = blockIdx.x * 64;
  int t  = threadIdx.x;
  const float* fs = Fs + (size_t)b * CC * NN;
  const float* ff = Ff + (size_t)b * CC * NN;
  float*       fl = Fl + (size_t)b * CC * NN;
  int nx = t & 63, cy = t >> 6;
#pragma unroll
  for (int i = 0; i < 16; ++i) {
    int c = cy + i * 4;
    float v = fs[(size_t)c * NN + n0 + nx] + ff[(size_t)c * NN + n0 + nx];
    fl[(size_t)c * NN + n0 + nx] = v;
    tile[c][nx] = v;
  }
  __syncthreads();
  h16* xo = xT + ((size_t)b * NN + n0) * CC;
  int cx = t & 63, ny = t >> 6;
#pragma unroll
  for (int i = 0; i < 16; ++i) {
    int n = ny + i * 4;
    xo[(size_t)n * CC + cx] = (h16)tile[cx][n];
  }
}

// ---------------------------------------------------------------------------
// Kernel 2: QKV projection GEMM. Each wave: 16 tokens x 320 outputs, K=64.
// V written transposed Vt[b][d][n].
// ---------------------------------------------------------------------------
__global__ void k_qkv_proj(const h16* __restrict__ xT, const h16* __restrict__ WB,
                           h16* __restrict__ Q1o, h16* __restrict__ K1o,
                           h16* __restrict__ Q2o, h16* __restrict__ K2o,
                           h16* __restrict__ Vt) {
  int lane = threadIdx.x & 31;
  int wave = threadIdx.x >> 5;
  int b    = blockIdx.y;
  int tok0 = blockIdx.x * 64 + wave * 16;
  const h16* xa = xT + ((size_t)b * NN + tok0) * CC;
  v16h a0 = load_a_frag(xa,      CC, lane);
  v16h a1 = load_a_frag(xa + 32, CC, lane);
  int rofs = (lane >> 4) * 8, col = lane & 15;
#pragma unroll
  for (int f = 0; f < 20; ++f) {
    v8f acc = zero8();
    acc = wmma16(a0, load_b_frag(WB + (size_t)(f * 16) * CC,      CC, lane), acc);
    acc = wmma16(a1, load_b_frag(WB + (size_t)(f * 16) * CC + 32, CC, lane), acc);
    int which = f >> 2, dsub = (f & 3) * 16;
    if (which < 4) {
      h16* dst;
      if      (which == 0) dst = Q1o;
      else if (which == 1) dst = K1o;
      else if (which == 2) dst = Q2o;
      else                 dst = K2o;
      dst += ((size_t)b * NN + tok0) * DD + dsub;
#pragma unroll
      for (int i = 0; i < 8; ++i)
        dst[(size_t)(i + rofs) * DD + col] = (h16)acc[i];
    } else {
      h16* dst = Vt + ((size_t)b * DD + dsub + col) * NN + tok0;
#pragma unroll
      for (int i = 0; i < 8; ++i)
        dst[i + rofs] = (h16)acc[i];
    }
  }
}

// ---------------------------------------------------------------------------
// Kernel 3: flash differential attention with async double-buffered LDS
// staging of the key tiles. Block = 256 threads (8 waves, 128 queries).
// Per 64-key tile the block stages K1 (8KB) + K2 (8KB) + V^T (8KB) once;
// all 8 waves consume B-fragments from LDS.
// ---------------------------------------------------------------------------
#define TILE_HALVES (64 * 64)          // one 64x64 f16 tile
#define STAGE_HALVES (3 * TILE_HALVES) // K1 + K2 + Vt

__device__ __forceinline__ void issue_stage(const h16* K1b, const h16* K2b,
                                            const h16* Vtb, int kt,
                                            h16* st, int t) {
#pragma unroll
  for (int u = 0; u < 6; ++u) {
    int ch  = t + u * 256;        // 0..1535 chunk of 8 halves
    int arr = ch >> 9;            // 0:K1 1:K2 2:Vt
    int rem = ch & 511;
    int r   = rem >> 3;           // row in tile
    int c8  = (rem & 7) * 8;      // col offset (halves)
    const h16* g;
    if      (arr == 0) g = K1b + (size_t)(kt + r) * DD + c8;
    else if (arr == 1) g = K2b + (size_t)(kt + r) * DD + c8;
    else               g = Vtb + (size_t)r * NN + kt + c8;
    stage_chunk16(st + arr * TILE_HALVES + r * 64 + c8, g);
  }
}

__global__ void k_attention(const h16* __restrict__ Q1, const h16* __restrict__ K1,
                            const h16* __restrict__ Q2, const h16* __restrict__ K2,
                            const h16* __restrict__ Vt, const float* __restrict__ lam_p,
                            h16* __restrict__ O) {
  __shared__ __align__(32) h16 stage[2][STAGE_HALVES];   // 48 KB
  __shared__ __align__(32) h16 Plds[8][16 * 64];         // 16 KB
  int t    = threadIdx.x;
  int lane = t & 31;
  int wave = t >> 5;
  int b    = blockIdx.y;
  int q0   = blockIdx.x * 128 + wave * 16;
  float lam = lam_p[0];
  const float scale = 0.125f;   // 1/sqrt(64)

  const h16* K1b = K1 + (size_t)b * NN * DD;
  const h16* K2b = K2 + (size_t)b * NN * DD;
  const h16* Vtb = Vt + (size_t)b * DD * NN;

  const h16* q1p = Q1 + ((size_t)b * NN + q0) * DD;
  const h16* q2p = Q2 + ((size_t)b * NN + q0) * DD;
  v16h qa1_0 = load_a_frag(q1p,      DD, lane);
  v16h qa1_1 = load_a_frag(q1p + 32, DD, lane);
  v16h qa2_0 = load_a_frag(q2p,      DD, lane);
  v16h qa2_1 = load_a_frag(q2p + 32, DD, lane);

  v8f o_acc[4];
#pragma unroll
  for (int f = 0; f < 4; ++f) o_acc[f] = zero8();
  float m_i[8], l_i[8];
#pragma unroll
  for (int i = 0; i < 8; ++i) { m_i[i] = -1e30f; l_i[i] = 0.f; }

  h16* myP = &Plds[wave][0];
  int rofs = (lane >> 4) * 8, col = lane & 15;

  issue_stage(K1b, K2b, Vtb, 0, &stage[0][0], t);

  for (int kt = 0, it = 0; kt < NN; kt += 64, ++it) {
    int cur = it & 1;
    bool have_next = (kt + 64) < NN;
    if (have_next)
      issue_stage(K1b, K2b, Vtb, kt + 64, &stage[1 - cur][0], t);
    wait_stage(have_next);
    __syncthreads();            // staged tile visible to all waves

    const h16* k1t = &stage[cur][0];                 // [key][d]
    const h16* k2t = &stage[cur][TILE_HALVES];       // [key][d]
    const h16* vtt = &stage[cur][2 * TILE_HALVES];   // [d][key]

    v8f s[4];
#pragma unroll
    for (int f = 0; f < 4; ++f) {
      const h16* k1p = k1t + (f * 16) * 64;
      const h16* k2p = k2t + (f * 16) * 64;
      v8f s1 = zero8(), s2 = zero8();
      s1 = wmma16(qa1_0, load_b_frag(k1p,      64, lane), s1);
      s1 = wmma16(qa1_1, load_b_frag(k1p + 32, 64, lane), s1);
      s2 = wmma16(qa2_0, load_b_frag(k2p,      64, lane), s2);
      s2 = wmma16(qa2_1, load_b_frag(k2p + 32, 64, lane), s2);
#pragma unroll
      for (int i = 0; i < 8; ++i) s[f][i] = (s1[i] - lam * s2[i]) * scale;
    }
    // row max across the 16 lanes of each lane-half (rows split 0-7 / 8-15)
    float mnew[8];
#pragma unroll
    for (int i = 0; i < 8; ++i) {
      float mx = fmaxf(fmaxf(s[0][i], s[1][i]), fmaxf(s[2][i], s[3][i]));
      mx = fmaxf(mx, __shfl_xor(mx, 1, 32));
      mx = fmaxf(mx, __shfl_xor(mx, 2, 32));
      mx = fmaxf(mx, __shfl_xor(mx, 4, 32));
      mx = fmaxf(mx, __shfl_xor(mx, 8, 32));
      mnew[i] = fmaxf(m_i[i], mx);
    }
    // P = exp(S - m), stage to wave-private LDS, row sums
    float rs[8];
#pragma unroll
    for (int i = 0; i < 8; ++i) rs[i] = 0.f;
#pragma unroll
    for (int f = 0; f < 4; ++f) {
#pragma unroll
      for (int i = 0; i < 8; ++i) {
        float p = __expf(s[f][i] - mnew[i]);
        rs[i] += p;
        myP[(i + rofs) * 64 + f * 16 + col] = (h16)p;
      }
    }
#pragma unroll
    for (int i = 0; i < 8; ++i) {
      float r = rs[i];
      r += __shfl_xor(r, 1, 32);
      r += __shfl_xor(r, 2, 32);
      r += __shfl_xor(r, 4, 32);
      r += __shfl_xor(r, 8, 32);
      float alpha = __expf(m_i[i] - mnew[i]);
      l_i[i] = l_i[i] * alpha + r;
      m_i[i] = mnew[i];
#pragma unroll
      for (int f = 0; f < 4; ++f) o_acc[f][i] *= alpha;
    }
    // O += P * V   (LDS V^T tile gives contiguous-K B fragments)
    v16h pa0 = load_a_frag(myP,      64, lane);
    v16h pa1 = load_a_frag(myP + 32, 64, lane);
#pragma unroll
    for (int f = 0; f < 4; ++f) {
      const h16* vp = vtt + (f * 16) * 64;
      o_acc[f] = wmma16(pa0, load_b_frag(vp,      64, lane), o_acc[f]);
      o_acc[f] = wmma16(pa1, load_b_frag(vp + 32, 64, lane), o_acc[f]);
    }
    __syncthreads();            // everyone done with stage[cur] before reuse
  }
  // normalize and store O [b][n][d] f16
  h16* op = O + ((size_t)b * NN + q0) * DD;
#pragma unroll
  for (int f = 0; f < 4; ++f) {
#pragma unroll
    for (int i = 0; i < 8; ++i)
      op[(size_t)(i + rofs) * DD + f * 16 + col] = (h16)(o_acc[f][i] / l_i[i]);
  }
}

// ---------------------------------------------------------------------------
// Kernel 4: fused proj (D->C) -> FFN1+GELU (C->HID) -> FFN2 (HID->C),
// y stored token-major fp32 + per-channel sum/sumsq atomics for BN.
// ---------------------------------------------------------------------------
__global__ void k_proj_ffn(const h16* __restrict__ O, const h16* __restrict__ Wph,
                           const h16* __restrict__ W1h, const h16* __restrict__ W2h,
                           float* __restrict__ Y, float* __restrict__ ssum,
                           float* __restrict__ ssq) {
  __shared__ __align__(32) h16 oT[4][16 * 64];
  __shared__ __align__(32) h16 hT[4][16 * 256];
  int lane = threadIdx.x & 31;
  int wave = threadIdx.x >> 5;
  int b    = blockIdx.y;
  int tok0 = blockIdx.x * 64 + wave * 16;
  int rofs = (lane >> 4) * 8, col = lane & 15;

  const h16* oa = O + ((size_t)b * NN + tok0) * DD;
  v16h a0 = load_a_frag(oa,      DD, lane);
  v16h a1 = load_a_frag(oa + 32, DD, lane);

  // o = Oattn * Wproj^T   (16 x 64)
  h16* myO = &oT[wave][0];
#pragma unroll
  for (int f = 0; f < 4; ++f) {
    v8f acc = zero8();
    acc = wmma16(a0, load_b_frag(Wph + (size_t)(f * 16) * DD,      DD, lane), acc);
    acc = wmma16(a1, load_b_frag(Wph + (size_t)(f * 16) * DD + 32, DD, lane), acc);
#pragma unroll
    for (int i = 0; i < 8; ++i)
      myO[(i + rofs) * 64 + f * 16 + col] = (h16)acc[i];
  }

  // h = GELU(o * W1^T)   (16 x 256)
  v16h oa0 = load_a_frag(myO,      64, lane);
  v16h oa1 = load_a_frag(myO + 32, 64, lane);
  h16* myH = &hT[wave][0];
#pragma unroll
  for (int f = 0; f < 16; ++f) {
    v8f acc = zero8();
    acc = wmma16(oa0, load_b_frag(W1h + (size_t)(f * 16) * CC,      CC, lane), acc);
    acc = wmma16(oa1, load_b_frag(W1h + (size_t)(f * 16) * CC + 32, CC, lane), acc);
#pragma unroll
    for (int i = 0; i < 8; ++i) {
      float x = acc[i];
      float g = 0.5f * x * (1.0f + erff(x * 0.70710678118654752f));
      myH[(i + rofs) * 256 + f * 16 + col] = (h16)g;
    }
  }

  // y = h * W2^T   (16 x 64), K = 256
  v8f y[4];
#pragma unroll
  for (int f = 0; f < 4; ++f) y[f] = zero8();
#pragma unroll
  for (int k = 0; k < 8; ++k) {
    v16h ha = load_a_frag(myH + k * 32, 256, lane);
#pragma unroll
    for (int f = 0; f < 4; ++f)
      y[f] = wmma16(ha, load_b_frag(W2h + (size_t)(f * 16) * HID + k * 32, HID, lane), y[f]);
  }

  float* yp = Y + ((size_t)b * NN + tok0) * CC;
#pragma unroll
  for (int f = 0; f < 4; ++f) {
    float ps = 0.f, pq = 0.f;
#pragma unroll
    for (int i = 0; i < 8; ++i) {
      float v = y[f][i];
      yp[(size_t)(i + rofs) * CC + f * 16 + col] = v;
      ps += v;
      pq += v * v;
    }
    ps += __shfl_xor(ps, 16, 32);   // fold rows 8..15 onto lanes 0..15
    pq += __shfl_xor(pq, 16, 32);
    if (lane < 16) {
      atomicAdd(&ssum[f * 16 + lane], ps);
      atomicAdd(&ssq [f * 16 + lane], pq);
    }
  }
}

// ---------------------------------------------------------------------------
// Kernel 5: BatchNorm finalize + residual, transpose [N][C] -> [C][N]
// ---------------------------------------------------------------------------
__global__ void k_bn_residual(const float* __restrict__ Y, const float* __restrict__ Fl,
                              const float* __restrict__ ssum, const float* __restrict__ ssq,
                              const float* __restrict__ gamma, const float* __restrict__ beta,
                              float* __restrict__ out) {
  __shared__ float tile[64][65];
  int b  = blockIdx.y;
  int n0 = blockIdx.x * 64;
  int t  = threadIdx.x;
  const float* yp = Y + ((size_t)b * NN + n0) * CC;
  int cx = t & 63, ny = t >> 6;
#pragma unroll
  for (int i = 0; i < 16; ++i) {
    int n = ny + i * 4;
    tile[n][cx] = yp[(size_t)n * CC + cx];
  }
  __syncthreads();
  const float inv = 1.0f / (float)(BB * NN);
  float*       op  = out + (size_t)b * CC * NN;
  const float* flp = Fl  + (size_t)b * CC * NN;
  int nx = t & 63, cy = t >> 6;
#pragma unroll
  for (int i = 0; i < 16; ++i) {
    int c = cy + i * 4;
    float mean = ssum[c] * inv;
    float var  = ssq[c] * inv - mean * mean;
    float rstd = rsqrtf(var + 1e-5f);
    float yv = (tile[nx][c] - mean) * rstd * gamma[c] + beta[c];
    op[(size_t)c * NN + n0 + nx] = yv + flp[(size_t)c * NN + n0 + nx];
  }
}

// ---------------------------------------------------------------------------
// Launch
// ---------------------------------------------------------------------------
extern "C" void kernel_launch(void* const* d_in, const int* in_sizes, int n_in,
                              void* d_out, int out_size, void* d_ws, size_t ws_size,
                              hipStream_t stream) {
  const float* Fs  = (const float*)d_in[0];
  const float* Ff  = (const float*)d_in[1];
  const float* Wq1 = (const float*)d_in[2];
  const float* Wk1 = (const float*)d_in[3];
  const float* Wq2 = (const float*)d_in[4];
  const float* Wk2 = (const float*)d_in[5];
  const float* Wv  = (const float*)d_in[6];
  const float* Wp  = (const float*)d_in[7];
  const float* W1  = (const float*)d_in[8];
  const float* W2  = (const float*)d_in[9];
  const float* gamma = (const float*)d_in[10];
  const float* beta  = (const float*)d_in[11];
  const float* lam   = (const float*)d_in[12];
  float* out = (float*)d_out;

  char* ws = (char*)d_ws;
  float* Fl  = (float*)(ws + 0);                  // 4 MB
  h16*   xT  = (h16*)  (ws + 4194304);            // 2 MB
  h16*   Q1  = (h16*)  (ws + 6291456);            // 2 MB
  h16*   K1h = (h16*)  (ws + 8388608);            // 2 MB
  h16*   Q2  = (h16*)  (ws + 10485760);           // 2 MB
  h16*   K2h = (h16*)  (ws + 12582912);           // 2 MB
  h16*   Vt  = (h16*)  (ws + 14680064);           // 2 MB
  h16*   Oat = (h16*)  (ws + 16777216);           // 2 MB
  float* Y   = (float*)(ws + 18874368);           // 4 MB
  h16*   WB  = (h16*)  (ws + 23068672);           // 40 KB
  h16*   Wph = (h16*)  (ws + 23109632);           // 8 KB
  h16*   W1h = (h16*)  (ws + 23117824);           // 32 KB
  h16*   W2h = (h16*)  (ws + 23150592);           // 32 KB
  float* ssum = (float*)(ws + 23183360);          // 64 f
  float* ssq  = (float*)(ws + 23183360 + 256);    // 64 f

  (void)hipMemsetAsync(ssum, 0, 256, stream);
  (void)hipMemsetAsync(ssq,  0, 256, stream);

  k_convert_weights<<<dim3(80), dim3(256), 0, stream>>>(
      Wq1, Wk1, Wq2, Wk2, Wv, Wp, W1, W2, WB, Wph, W1h, W2h);

  k_add_transpose<<<dim3(NN / 64, BB), dim3(256), 0, stream>>>(Fs, Ff, Fl, xT);

  k_qkv_proj<<<dim3(NN / 64, BB), dim3(128), 0, stream>>>(
      xT, WB, Q1, K1h, Q2, K2h, Vt);

  k_attention<<<dim3(NN / 128, BB), dim3(256), 0, stream>>>(
      Q1, K1h, Q2, K2h, Vt, lam, Oat);

  k_proj_ffn<<<dim3(NN / 64, BB), dim3(128), 0, stream>>>(
      Oat, Wph, W1h, W2h, Y, ssum, ssq);

  k_bn_residual<<<dim3(NN / 64, BB), dim3(256), 0, stream>>>(
      Y, Fl, ssum, ssq, gamma, beta, out);
}
